// ShannonCapacityPooling_26010321945083
// MI455X (gfx1250) — matrix operand order repaired and checked by reading, not compiled, not run on hardware
//
#include <hip/hip_runtime.h>
#include <hip/hip_bf16.h>

// ---------------------------------------------------------------------------
// ShannonCapacityPooling for MI455X (gfx1250, wave32, WMMA)
//   B=16, N=512, F=64, HID=128, FE=4, THRESH=0.5
// GEMMs on WMMA: f32 16x16x4 for exact-f32 paths (gram, channel@prob, MLP),
// f16 16x16x32 for the binary adj@adj (exact: 0/1 inputs, counts <= 511),
// staged through LDS with GLOBAL_LOAD_ASYNC_TO_LDS_B128 (ASYNCcnt).
// Greedy MWIS: 1 workgroup/batch, adjacency bitmask in LDS.
// ---------------------------------------------------------------------------

#define B_   16
#define N_   512
#define F_   64
#define HID_ 128

typedef float    v2f  __attribute__((ext_vector_type(2)));
typedef float    v8f  __attribute__((ext_vector_type(8)));
typedef _Float16 v16h __attribute__((ext_vector_type(16)));

// workspace offsets (4-byte units)
#define WS_SQ      0         // B*N
#define WS_CHROW   8192      // B*N
#define WS_BUFP    16384     // B*N*HID
#define WS_AGG     1064960   // B*N*HID
#define WS_TBUF    2113536   // B*N*HID
#define WS_PROBV   3162112   // B*N
#define WS_LOSSTHR 3170304   // B
#define WS_GMAX    3170320   // 1 (uint bits)
#define WS_COUNT   3170321   // B ints
#define WS_PERM    3170337   // B*N ints

// output offsets (floats): x_pooled | adj_pooled | loss | sel(int32 bits)
#define OUT_X    0
#define OUT_ADJ  524288
#define OUT_LOSS 4718592
#define OUT_SEL  4718593

// ---------------- WMMA helpers ----------------

// D = A(16x4 f32) * B(4x16 f32) + C, wave32.
// A layout: lanes 0-15 hold M=lane {K=k0,k0+1}; lanes 16-31 hold {K=k0+2,k0+3}.
// B layout (transposed convention): lanes 0-15 hold N=lane {K=k0,k0+1}; 16-31 {k0+2,k0+3}.
__device__ __forceinline__ v8f wmma_f32_k4(v2f a, v2f b, v8f c, int lane) {
#if __has_builtin(__builtin_amdgcn_wmma_f32_16x16x4_f32)
  return __builtin_amdgcn_wmma_f32_16x16x4_f32(false, a, false, b, (short)0, c,
                                               false, false);
#else
  int lm = lane & 15;
#pragma unroll
  for (int r = 0; r < 8; ++r) {
    int m = (r + ((lane >> 4) << 3)) & 15;
#pragma unroll
    for (int k = 0; k < 4; ++k) {
      float av = __shfl((k & 1) ? a.y : a.x, m  | ((k >> 1) << 4), 32);
      float bv = __shfl((k & 1) ? b.y : b.x, lm | ((k >> 1) << 4), 32);
      c[r] += av * bv;
    }
  }
  return c;
#endif
}

__device__ __forceinline__ v8f wmma_f16_k32(v16h a, v16h b, v8f c) {
  return __builtin_amdgcn_wmma_f32_16x16x32_f16(false, a, false, b, (short)0, c,
                                                false, false);
}

// low 32 bits of a generic LDS pointer == byte offset in the wave's LDS space
__device__ __forceinline__ unsigned lds_addr32(const void* p) {
  return (unsigned)(unsigned long long)p;
}

// one-lane async copy of 16 bytes global -> LDS (tracked by ASYNCcnt)
__device__ __forceinline__ void async_g2lds_b128(unsigned lds_byte,
                                                 const float* gaddr) {
  asm volatile("global_load_async_to_lds_b128 %0, %1, off"
               :: "v"(lds_byte), "v"(gaddr)
               : "memory");
}

__device__ __forceinline__ void wait_async0() {
  asm volatile("s_wait_asynccnt 0x0" ::: "memory");
}

// ---------------- small kernels ----------------

__global__ void k_init(unsigned* gmax) { *gmax = 0u; }

__global__ void k_sq(const float* __restrict__ x, float* __restrict__ sq) {
  int idx = blockIdx.x * blockDim.x + threadIdx.x;
  if (idx >= B_ * N_) return;
  const float* xr = x + (size_t)idx * F_;
  float s = 0.f;
#pragma unroll 8
  for (int f = 0; f < F_; ++f) s += xr[f] * xr[f];
  sq[idx] = s;
}

// gram via f32 WMMA -> dist*adj (raw channel) + global max (uint-bit atomicMax)
__global__ void k_gram_channel(const float* __restrict__ x,
                               const float* __restrict__ adj,
                               const float* __restrict__ sq,
                               float* __restrict__ chan,
                               unsigned* __restrict__ gmax) {
  int wid = threadIdx.x >> 5, lane = threadIdx.x & 31;
  int b = blockIdx.z;
  int m0 = blockIdx.y * 16;
  int n0 = (blockIdx.x * 4 + wid) * 16;
  int lm = lane & 15, hi = lane >> 4;
  const float* xb = x + (size_t)b * N_ * F_;
  v8f acc = {};
#pragma unroll 4
  for (int k0 = 0; k0 < F_; k0 += 4) {
    int k = k0 + 2 * hi;
    v2f a, bb;
    a.x  = xb[(size_t)(m0 + lm) * F_ + k];
    a.y  = xb[(size_t)(m0 + lm) * F_ + k + 1];
    bb.x = xb[(size_t)(n0 + lm) * F_ + k];
    bb.y = xb[(size_t)(n0 + lm) * F_ + k + 1];
    acc = wmma_f32_k4(a, bb, acc, lane);
  }
  const float* sqb  = sq + b * N_;
  const float* adjb = adj + (size_t)b * N_ * N_;
  float* chb = chan + (size_t)b * N_ * N_;
  float lmax = 0.f;
#pragma unroll
  for (int r = 0; r < 8; ++r) {
    int m = m0 + r + 8 * hi, n = n0 + lm;
    float d2 = sqb[m] + sqb[n] - 2.f * acc[r];
    float d = sqrtf(fmaxf(d2, 0.f));
    float raw = d * adjb[(size_t)m * N_ + n];
    chb[(size_t)m * N_ + n] = raw;
    lmax = fmaxf(lmax, raw);
  }
  __shared__ float smax[128];
  smax[threadIdx.x] = lmax;
  __syncthreads();
  for (int off = 64; off; off >>= 1) {
    if (threadIdx.x < off)
      smax[threadIdx.x] = fmaxf(smax[threadIdx.x], smax[threadIdx.x + off]);
    __syncthreads();
  }
  if (threadIdx.x == 0) atomicMax(gmax, __float_as_uint(smax[0]));
}

// channel = (0 < raw <= 0.5*max) ? 1 : 0 ; also row sums for GIN layer 1
__global__ void k_binarize_rowsum(float* __restrict__ chan,
                                  const unsigned* __restrict__ gmax,
                                  float* __restrict__ chrow) {
  int row = blockIdx.x;  // b*N + n
  float thr = 0.5f * __uint_as_float(*gmax);
  float* cr = chan + (size_t)row * N_;
  float s = 0.f;
  for (int j = threadIdx.x; j < N_; j += 128) {
    float c = cr[j];
    float kb = (c > 0.f && c <= thr) ? 1.f : 0.f;
    cr[j] = kb;
    s += kb;
  }
  __shared__ float red[128];
  red[threadIdx.x] = s;
  __syncthreads();
  for (int off = 64; off; off >>= 1) {
    if (threadIdx.x < off) red[threadIdx.x] += red[threadIdx.x + off];
    __syncthreads();
  }
  if (threadIdx.x == 0) chrow[row] = red[0];
}

// GIN layer 1: h0 = rowsum+1 (scalar); prob1 = relu(relu(h0*W1a+b1a)@W1b+b1b)
__global__ void k_layer1(const float* __restrict__ chrow,
                         const float* __restrict__ W1a, const float* __restrict__ b1a,
                         const float* __restrict__ W1b, const float* __restrict__ b1b,
                         float* __restrict__ out) {
  int node = blockIdx.x;   // b*N+n
  int j = threadIdx.x;     // 0..127
  float h0 = chrow[node] + 1.f;
  float s = b1b[j];
#pragma unroll
  for (int e = 0; e < 4; ++e) {
    float z = fmaxf(h0 * W1a[e] + b1a[e], 0.f);
    s += z * W1b[e * HID_ + j];
  }
  out[(size_t)node * HID_ + j] = fmaxf(s, 0.f);
}

// agg = channel @ pin + pin  (f32 WMMA, [512x512]*[512x128] per batch)
__global__ void k_agg(const float* __restrict__ chan,
                      const float* __restrict__ pin, float* __restrict__ out) {
  int wid = threadIdx.x >> 5, lane = threadIdx.x & 31;
  int b = blockIdx.z;
  int m0 = blockIdx.y * 16;
  int n0 = (blockIdx.x * 4 + wid) * 16;
  int lm = lane & 15, hi = lane >> 4;
  const float* chb = chan + (size_t)b * N_ * N_;
  const float* pb  = pin  + (size_t)b * N_ * HID_;
  __builtin_prefetch(chb + (size_t)(m0 + lm) * N_, 0, 1);
  v8f acc = {};
  for (int k0 = 0; k0 < N_; k0 += 4) {
    int k = k0 + 2 * hi;
    v2f a, bb;
    a.x  = chb[(size_t)(m0 + lm) * N_ + k];
    a.y  = chb[(size_t)(m0 + lm) * N_ + k + 1];
    bb.x = pb[(size_t)k * HID_ + n0 + lm];
    bb.y = pb[(size_t)(k + 1) * HID_ + n0 + lm];
    acc = wmma_f32_k4(a, bb, acc, lane);
  }
  float* ob = out + (size_t)b * N_ * HID_;
#pragma unroll
  for (int r = 0; r < 8; ++r) {
    int m = m0 + r + 8 * hi, n = n0 + lm;
    ob[(size_t)m * HID_ + n] = acc[r] + pb[(size_t)m * HID_ + n];
  }
}

// out = act(in @ W + bias), W shared across batches ([512x128]*[128x128])
__global__ void k_mlp(const float* __restrict__ in, const float* __restrict__ W,
                      const float* __restrict__ bias, float* __restrict__ out,
                      int relu) {
  int wid = threadIdx.x >> 5, lane = threadIdx.x & 31;
  int b = blockIdx.z;
  int m0 = blockIdx.y * 16;
  int n0 = (blockIdx.x * 4 + wid) * 16;
  int lm = lane & 15, hi = lane >> 4;
  const float* inb = in + (size_t)b * N_ * HID_;
  v8f acc = {};
#pragma unroll 4
  for (int k0 = 0; k0 < HID_; k0 += 4) {
    int k = k0 + 2 * hi;
    v2f a, bb;
    a.x  = inb[(size_t)(m0 + lm) * HID_ + k];
    a.y  = inb[(size_t)(m0 + lm) * HID_ + k + 1];
    bb.x = W[(size_t)k * HID_ + n0 + lm];
    bb.y = W[(size_t)(k + 1) * HID_ + n0 + lm];
    acc = wmma_f32_k4(a, bb, acc, lane);
  }
  float* ob = out + (size_t)b * N_ * HID_;
#pragma unroll
  for (int r = 0; r < 8; ++r) {
    int m = m0 + r + 8 * hi, n = n0 + lm;
    float v = acc[r] + bias[n];
    if (relu) v = fmaxf(v, 0.f);
    ob[(size_t)m * HID_ + n] = v;
  }
}

// prob = sigmoid(t @ W3b + b3b)
__global__ void k_final(const float* __restrict__ t, const float* __restrict__ W3b,
                        const float* __restrict__ b3b, float* __restrict__ probv) {
  int idx = blockIdx.x * blockDim.x + threadIdx.x;
  if (idx >= B_ * N_) return;
  const float* tr = t + (size_t)idx * HID_;
  float s = b3b[0];
#pragma unroll 8
  for (int j = 0; j < HID_; ++j) s += tr[j] * W3b[j];
  probv[idx] = 1.f / (1.f + expf(-s));
}

// loss_thresholds[b] = N - sum(p) + p^T adj p
__global__ void k_loss(const float* __restrict__ adj,
                       const float* __restrict__ probv,
                       float* __restrict__ lossthr) {
  int b = blockIdx.x, tid = threadIdx.x;
  __shared__ float pl[N_];
  __shared__ float r1[256], r2[256];
  for (int r = tid; r < N_; r += 256) pl[r] = probv[b * N_ + r];
  __syncthreads();
  const float* ab = adj + (size_t)b * N_ * N_;
  float s1 = 0.f, s2 = 0.f;
  for (int r = tid; r < N_; r += 256) {
    const float* arow = ab + (size_t)r * N_;
    float y = 0.f;
    for (int j = 0; j < N_; ++j) y += arow[j] * pl[j];
    s1 += pl[r];
    s2 += pl[r] * y;
  }
  r1[tid] = s1; r2[tid] = s2;
  __syncthreads();
  for (int off = 128; off; off >>= 1) {
    if (tid < off) { r1[tid] += r1[tid + off]; r2[tid] += r2[tid + off]; }
    __syncthreads();
  }
  if (tid == 0) lossthr[b] = (float)N_ - r1[0] + r2[0];
}

__global__ void k_loss_sum(const float* __restrict__ lt, float* __restrict__ out) {
  float s = 0.f;
  for (int i = 0; i < B_; ++i) s += lt[i];
  *out = s / (float)B_;
}

// sequential greedy MWIS; one block per batch; adjacency as LDS bitmask
__global__ void __launch_bounds__(256)
k_greedy(const float* __restrict__ adj, const float* __restrict__ probv,
         const float* __restrict__ lossthr, int* __restrict__ selOut,
         int* __restrict__ permG, int* __restrict__ countG) {
  __shared__ unsigned adjb[N_ * 16];       // 32 KB bitmask
  __shared__ float pL[N_], tempL[N_];
  __shared__ int orderL[N_];
  __shared__ unsigned short degL[N_];
  __shared__ unsigned char selL[N_], rejL[N_];
  __shared__ float red1[256], red2[256];
  __shared__ int flagL, countL;
  __shared__ float thrL;

  int b = blockIdx.x, tid = threadIdx.x;
  const float* ab = adj + (size_t)b * N_ * N_;

  for (int r = tid; r < N_; r += 256) {
    int deg = 0;
    const float* arow = ab + (size_t)r * N_;
#pragma unroll
    for (int w = 0; w < 16; ++w) {
      unsigned bits = 0;
#pragma unroll
      for (int t = 0; t < 32; ++t)
        bits |= (arow[w * 32 + t] > 0.f) ? (1u << t) : 0u;
      adjb[r * 16 + w] = bits;
      deg += __popc(bits);
    }
    degL[r] = (unsigned short)deg;
    pL[r] = probv[b * N_ + r];
    selL[r] = 0; rejL[r] = 0;
  }
  if (tid == 0) thrL = lossthr[b];
  __syncthreads();

  // stable descending rank sort (argsort(-p), ties by index ascending)
  for (int r = tid; r < N_; r += 256) {
    float pr = pL[r];
    int cnt = 0;
    for (int j = 0; j < N_; ++j) {
      float pj = pL[j];
      cnt += (pj > pr) || (pj == pr && j < r);
    }
    orderL[cnt] = r;
  }
  __syncthreads();

  for (int it = 0; it < N_; ++it) {
    int node = orderL[it];
    float s1 = 0.f;
    for (int r = tid; r < N_; r += 256) {
      int nb = (adjb[node * 16 + (r >> 5)] >> (r & 31)) & 1;
      float t = (r == node) ? 1.f : (nb ? 0.f : pL[r]);
      tempL[r] = t;
      s1 += t;
    }
    red1[tid] = s1;
    __syncthreads();
    float s2 = 0.f;
    for (int r = tid; r < N_; r += 256) {
      float ti = tempL[r];
      if (ti != 0.f) {
        float acc = 0.f;
#pragma unroll
        for (int w = 0; w < 16; ++w) {
          unsigned mm = adjb[r * 16 + w];
          while (mm) {
            int j = __ffs(mm) - 1;
            acc += tempL[w * 32 + j];
            mm &= mm - 1;
          }
        }
        s2 += ti * acc;
      }
    }
    red2[tid] = s2;
    __syncthreads();
    for (int off = 128; off; off >>= 1) {
      if (tid < off) { red1[tid] += red1[tid + off]; red2[tid] += red2[tid + off]; }
      __syncthreads();
    }
    if (tid == 0) {
      float lo = (float)N_ - red1[0] + red2[0];
      int deg = degL[node];
      int accept = (deg > 0) && !rejL[node] && !selL[node] && (lo <= thrL);
      if (deg == 0 || accept) selL[node] = 1;
      flagL = accept;
    }
    __syncthreads();
    if (flagL) {
      for (int r = tid; r < N_; r += 256) {
        if ((adjb[node * 16 + (r >> 5)] >> (r & 31)) & 1) rejL[r] = 1;
        pL[r] = tempL[r];
      }
    }
    __syncthreads();
  }

  float c1 = 0.f;
  for (int r = tid; r < N_; r += 256) c1 += (float)selL[r];
  red1[tid] = c1;
  __syncthreads();
  for (int off = 128; off; off >>= 1) {
    if (tid < off) red1[tid] += red1[tid + off];
    __syncthreads();
  }
  if (tid == 0) { countL = (int)red1[0]; countG[b] = countL; }
  __syncthreads();

  for (int r = tid; r < N_; r += 256) {
    int pre = 0;
    for (int j = 0; j < r; ++j) pre += selL[j];
    int rank = selL[r] ? pre : (countL + r - pre);
    permG[b * N_ + rank] = r;
    selOut[b * N_ + r] = (int)selL[r];
  }
}

__global__ void k_xpool(const float* __restrict__ x, const int* __restrict__ perm,
                        const int* __restrict__ count, float* __restrict__ out) {
  int b = blockIdx.y, i = blockIdx.x, f = threadIdx.x;
  int cnt = count[b];
  int src = perm[b * N_ + i];
  float v = (i < cnt) ? x[(size_t)(b * N_ + src) * F_ + f] : 0.f;
  out[(size_t)(b * N_ + i) * F_ + f] = v;
}

// adj_pooled = mask * clip((adj@adj)[perm][:,perm] - diag, 0, 1)
//   C[i][j] = rowdot(adj[perm_i], adj[perm_j]) via f16 WMMA (exact for 0/1).
// Block = 4 waves: 16 output rows x 64 output cols. perm-gathered A (16 rows)
// and B (64 rows) slabs staged into LDS per 128-wide K chunk with
// GLOBAL_LOAD_ASYNC_TO_LDS_B128 (ASYNCcnt), fragments read back from LDS.
__global__ void __launch_bounds__(128)
k_adjpool(const float* __restrict__ adj, const int* __restrict__ perm,
          const int* __restrict__ count, float* __restrict__ out) {
  __shared__ __align__(16) float As[16 * 128];   // 8 KB
  __shared__ __align__(16) float Bs[64 * 128];   // 32 KB
  int wid = threadIdx.x >> 5, lane = threadIdx.x & 31;
  int b = blockIdx.z;
  int m0 = blockIdx.y * 16;          // 32 m-tiles
  int nbase = blockIdx.x * 64;       // 8 n-macroblocks
  int n0 = nbase + wid * 16;
  int lm = lane & 15, hi = lane >> 4;
  int cnt = count[b];
  const int* pb = perm + b * N_;
  const float* ab = adj + (size_t)b * N_ * N_;
  unsigned asBase = lds_addr32(As);
  unsigned bsBase = lds_addr32(Bs);

  int abase = hi ? 8 : 0;   // A frag K sub-offsets (16-bit A 16x32 layout)
  int bbase = hi ? 8 : 0;   // B frag K sub-offsets (16-bit B 32x16 layout)
  v8f acc = {};

  for (int c = 0; c < 4; ++c) {
    int k0c = c * 128;
    // stage A slab: 16 rows x 128 f32 = 512 x b128, 4 per thread
#pragma unroll
    for (int t = 0; t < 4; ++t) {
      int idx = threadIdx.x + t * 128;   // 0..511
      int row = idx >> 5, c4 = idx & 31;
      const float* g = ab + (size_t)pb[m0 + row] * N_ + k0c + c4 * 4;
      unsigned l = asBase + (unsigned)(row * 128 + c4 * 4) * 4u;
      async_g2lds_b128(l, g);
    }
    // stage B slab: 64 rows x 128 f32 = 2048 x b128, 16 per thread
#pragma unroll
    for (int t = 0; t < 16; ++t) {
      int idx = threadIdx.x + t * 128;   // 0..2047
      int row = idx >> 5, c4 = idx & 31;
      const float* g = ab + (size_t)pb[nbase + row] * N_ + k0c + c4 * 4;
      unsigned l = bsBase + (unsigned)(row * 128 + c4 * 4) * 4u;
      async_g2lds_b128(l, g);
    }
    wait_async0();
    __syncthreads();

#pragma unroll
    for (int kk = 0; kk < 128; kk += 32) {
      v16h af, bf;
#pragma unroll
      for (int v = 0; v < 8; ++v) {
        int ka = kk + abase + ((v < 4) ? 2 * v : 16 + 2 * (v - 4));
        af[2 * v]     = (_Float16)As[lm * 128 + ka];
        af[2 * v + 1] = (_Float16)As[lm * 128 + ka + 1];
        int kb = kk + bbase + ((v < 4) ? 2 * v : 16 + 2 * (v - 4));
        bf[2 * v]     = (_Float16)Bs[(wid * 16 + lm) * 128 + kb];
        bf[2 * v + 1] = (_Float16)Bs[(wid * 16 + lm) * 128 + kb + 1];
      }
      acc = wmma_f16_k32(af, bf, acc);
    }
    __syncthreads();   // protect slabs before next chunk overwrite
  }

#pragma unroll
  for (int r = 0; r < 8; ++r) {
    int m = m0 + r + 8 * hi, n = n0 + lm;
    float v = fminf(fmaxf(acc[r], 0.f), 1.f);
    if (m == n || m >= cnt || n >= cnt) v = 0.f;
    out[(size_t)(b * N_ + m) * N_ + n] = v;
  }
}

// ---------------------------------------------------------------------------

extern "C" void kernel_launch(void* const* d_in, const int* in_sizes, int n_in,
                              void* d_out, int out_size, void* d_ws, size_t ws_size,
                              hipStream_t stream) {
  const float* x   = (const float*)d_in[0];
  const float* adj = (const float*)d_in[1];
  const float* W1a = (const float*)d_in[2];
  const float* b1a = (const float*)d_in[3];
  const float* W1b = (const float*)d_in[4];
  const float* b1b = (const float*)d_in[5];
  const float* W2a = (const float*)d_in[6];
  const float* b2a = (const float*)d_in[7];
  const float* W2b = (const float*)d_in[8];
  const float* b2b = (const float*)d_in[9];
  const float* W3a = (const float*)d_in[10];
  const float* b3a = (const float*)d_in[11];
  const float* W3b = (const float*)d_in[12];
  const float* b3b = (const float*)d_in[13];

  float* outf = (float*)d_out;
  float* ws = (float*)d_ws;
  float* sq      = ws + WS_SQ;
  float* chrow   = ws + WS_CHROW;
  float* bufP    = ws + WS_BUFP;
  float* aggb    = ws + WS_AGG;
  float* tbuf    = ws + WS_TBUF;
  float* probv   = ws + WS_PROBV;
  float* lossthr = ws + WS_LOSSTHR;
  unsigned* gmax = (unsigned*)(ws + WS_GMAX);
  int* cnt  = (int*)(ws + WS_COUNT);
  int* perm = (int*)(ws + WS_PERM);
  // channel tensor lives in the adj_pooled output region until the final GEMM
  float* chan = outf + OUT_ADJ;
  int* selOut = (int*)(outf + OUT_SEL);

  k_init<<<1, 1, 0, stream>>>(gmax);
  k_sq<<<32, 256, 0, stream>>>(x, sq);
  k_gram_channel<<<dim3(8, 32, B_), 128, 0, stream>>>(x, adj, sq, chan, gmax);
  k_binarize_rowsum<<<B_ * N_, 128, 0, stream>>>(chan, gmax, chrow);
  k_layer1<<<B_ * N_, HID_, 0, stream>>>(chrow, W1a, b1a, W1b, b1b, bufP);
  k_agg<<<dim3(2, 32, B_), 128, 0, stream>>>(chan, bufP, aggb);
  k_mlp<<<dim3(2, 32, B_), 128, 0, stream>>>(aggb, W2a, b2a, tbuf, 1);
  k_mlp<<<dim3(2, 32, B_), 128, 0, stream>>>(tbuf, W2b, b2b, bufP, 1);
  k_agg<<<dim3(2, 32, B_), 128, 0, stream>>>(chan, bufP, aggb);
  k_mlp<<<dim3(2, 32, B_), 128, 0, stream>>>(aggb, W3a, b3a, tbuf, 1);
  k_final<<<32, 256, 0, stream>>>(tbuf, W3b, b3b, probv);
  k_loss<<<B_, 256, 0, stream>>>(adj, probv, lossthr);
  k_loss_sum<<<1, 1, 0, stream>>>(lossthr, outf + OUT_LOSS);
  k_greedy<<<B_, 256, 0, stream>>>(adj, probv, lossthr, selOut, perm, cnt);
  k_xpool<<<dim3(N_, B_), F_, 0, stream>>>(x, perm, cnt, outf + OUT_X);
  k_adjpool<<<dim3(8, 32, B_), 128, 0, stream>>>(adj, perm, cnt, outf + OUT_ADJ);
}
